// DynamicSubspaceUsage_7378753815114
// MI455X (gfx1250) — matrix operand order repaired and checked by reading, not compiled, not run on hardware
//
#include <hip/hip_runtime.h>

#define D_MODEL 1024
#define H_DIM   512
#define K_SEL   614
#define M_TILE  16
#define BLOCK_THREADS 256
#define N_ROWS (8 * 4096)

typedef __attribute__((ext_vector_type(16))) __bf16        v16bf;
typedef __attribute__((ext_vector_type(8)))  float         v8f;
typedef __attribute__((ext_vector_type(8)))  unsigned int  v8u;
typedef __attribute__((ext_vector_type(4)))  unsigned int  v4u;

static __device__ __forceinline__ unsigned short f32_to_bf16(float f) {
    unsigned u = __float_as_uint(f);
    unsigned rnd = 0x7FFFu + ((u >> 16) & 1u);   // round-to-nearest-even
    u += rnd;
    return (unsigned short)(u >> 16);
}

// 32-byte fragment load as two 16B vectors (works for LDS and global pointers)
static __device__ __forceinline__ v8u load_frag(const unsigned short* p) {
    const v4u* q = (const v4u*)p;
    v4u lo = q[0];
    v4u hi = q[1];
    v8u r = {lo.x, lo.y, lo.z, lo.w, hi.x, hi.y, hi.z, hi.w};
    return r;
}

// --------------------------------------------------------------------------
// Pre-pack weights into B-fragment-ready bf16 blocks:
// block (kb, n) holds W[kb*32 + ko][n], ko = 0..31, contiguous (64 bytes).
// Lane L<16 reads bytes [0,16),[32,48) (K 0..7,16..23), lane>=16 reads [16,32),[48,64) (K 8..15,24..31);
// within each half, K-pairs are ascending -> matches 16-bit B VGPR layout.
// --------------------------------------------------------------------------
__global__ __launch_bounds__(256) void pack_weights_kernel(
    const float* __restrict__ w1, const float* __restrict__ w2,
    unsigned short* __restrict__ w1p, unsigned short* __restrict__ w2p)
{
    int idx = blockIdx.x * blockDim.x + threadIdx.x;
    const int total1 = D_MODEL * H_DIM;
    const int total2 = H_DIM * D_MODEL;
    if (idx < total1) {
        int k = idx / H_DIM, n = idx % H_DIM;
        int kb = k >> 5, ko = k & 31;
        w1p[(((size_t)kb * H_DIM + n) << 5) + ko] = f32_to_bf16(w1[idx]);
    } else if (idx < total1 + total2) {
        int j = idx - total1;
        int k = j / D_MODEL, n = j % D_MODEL;
        int kb = k >> 5, ko = k & 31;
        w2p[(((size_t)kb * D_MODEL + n) << 5) + ko] = f32_to_bf16(w2[j]);
    }
}

// --------------------------------------------------------------------------
// Fused: GEMM1(+bias,ReLU) -> GEMM2(+bias) -> per-row exact top-K mask.
// One block = 16 rows (one WMMA M-tile), 8 waves of 32.
// --------------------------------------------------------------------------
__global__ __launch_bounds__(BLOCK_THREADS) void fused_topk_mask_kernel(
    const float* __restrict__ emb,
    const float* __restrict__ b1,
    const float* __restrict__ b2,
    const unsigned short* __restrict__ w1p,
    const unsigned short* __restrict__ w2p,
    float* __restrict__ out)
{
    extern __shared__ unsigned short smem[];
    unsigned short* XA = smem;                         // 32 KB: 32 kb * 32 lanes * 16 bf16
    unsigned short* HA = smem + 32 * 512;              // 16 KB: 16 kb * 32 lanes * 16 bf16
    float*          Lf = (float*)(smem + 48 * 512);    // 64 KB: 16 rows * 1024 f32 logits

    const int tid  = threadIdx.x;
    const int lane = tid & 31;
    const int wv   = tid >> 5;
    const int ln   = lane & 15;      // N index within tile
    const int lh   = lane >> 4;      // half select (K-halves / M-halves)
    const long row0 = (long)blockIdx.x * M_TILE;

    // ---- Phase 0: load fp32 X tile, convert to bf16 A-fragment layout ----
    // 16-bit A 16x32: lane<16 holds M=lane, K={0..7,16..23}; lane>=16 holds
    // M=lane-16, K={8..15,24..31}; pairs packed K-ascending per dword.
    for (int i = tid; i < M_TILE * D_MODEL; i += BLOCK_THREADS) {
        int m = i >> 10;
        int k = i & (D_MODEL - 1);
        float x = emb[(row0 + m) * D_MODEL + k];
        int kb = k >> 5, ko = k & 31;
        int l2, s2;
        if      (ko < 8)  { l2 = m;      s2 = ko;      }
        else if (ko < 16) { l2 = m + 16; s2 = ko - 8;  }
        else if (ko < 24) { l2 = m;      s2 = ko - 8;  }
        else              { l2 = m + 16; s2 = ko - 16; }
        XA[(kb << 9) + (l2 << 4) + s2] = f32_to_bf16(x);
    }
    __syncthreads();

    // ---- Phase 1: H = relu(X @ W1 + b1), stored as bf16 A-fragments ----
    for (int nt = wv; nt < H_DIM / 16; nt += 8) {
        int n0 = nt * 16;
        v8f acc = {};
        for (int kb = 0; kb < D_MODEL / 32; ++kb) {
            v8u ar = load_frag(XA + (kb << 9) + (lane << 4));
            const v4u* bq = (const v4u*)(w1p + (((size_t)kb * H_DIM + n0 + ln) << 5));
            v4u blo = bq[lh], bhi = bq[2 + lh];
            v8u br = {blo.x, blo.y, blo.z, blo.w, bhi.x, bhi.y, bhi.z, bhi.w};
            acc = __builtin_amdgcn_wmma_f32_16x16x32_bf16(
                false, __builtin_bit_cast(v16bf, ar),
                false, __builtin_bit_cast(v16bf, br),
                (short)0, acc, false, false);
        }
        float bias = b1[n0 + ln];
        // C layout: vgpr r holds (M = r + 8*lh, N = n0 + ln). Scatter into
        // A-fragment layout for GEMM2 with K2 = n0 + ln.
        int k2 = n0 + ln;
        int kb2 = k2 >> 5, ko = k2 & 31;
        int s2, ladd;
        if      (ko < 8)  { s2 = ko;      ladd = 0;  }
        else if (ko < 16) { s2 = ko - 8;  ladd = 16; }
        else if (ko < 24) { s2 = ko - 8;  ladd = 0;  }
        else              { s2 = ko - 16; ladd = 16; }
#pragma unroll
        for (int r = 0; r < 8; ++r) {
            int m = r + 8 * lh;
            float h = fmaxf(acc[r] + bias, 0.0f);
            HA[(kb2 << 9) + ((m + ladd) << 4) + s2] = f32_to_bf16(h);
        }
    }
    __syncthreads();

    // ---- Phase 2: logits = H @ W2 + b2 (fp32, stays in LDS) ----
    // sigmoid is monotone -> selection on raw logits == selection on sigmoid.
    for (int nt = wv; nt < D_MODEL / 16; nt += 8) {
        int n0 = nt * 16;
        v8f acc = {};
        for (int kb = 0; kb < H_DIM / 32; ++kb) {
            v8u ar = load_frag(HA + (kb << 9) + (lane << 4));
            const v4u* bq = (const v4u*)(w2p + (((size_t)kb * D_MODEL + n0 + ln) << 5));
            v4u blo = bq[lh], bhi = bq[2 + lh];
            v8u br = {blo.x, blo.y, blo.z, blo.w, bhi.x, bhi.y, bhi.z, bhi.w};
            acc = __builtin_amdgcn_wmma_f32_16x16x32_bf16(
                false, __builtin_bit_cast(v16bf, ar),
                false, __builtin_bit_cast(v16bf, br),
                (short)0, acc, false, false);
        }
        float bias = b2[n0 + ln];
#pragma unroll
        for (int r = 0; r < 8; ++r) {
            int m = r + 8 * lh;
            Lf[m * D_MODEL + n0 + ln] = acc[r] + bias;
        }
    }
    __syncthreads();

    // ---- Phase 3: exact per-row top-K threshold + masked output ----
    // One wave handles two rows. 32-bit radix select on order-preserving
    // uint keys finds the exact K-th-largest value; ties resolved by index
    // (lowest index first, matching jax.lax.top_k stability).
    for (int rr = 0; rr < 2; ++rr) {
        int m = wv * 2 + rr;
        unsigned key[32];
#pragma unroll
        for (int j = 0; j < 32; ++j) {
            unsigned u = __float_as_uint(Lf[m * D_MODEL + j * 32 + lane]);
            key[j] = u ^ ((unsigned)((int)u >> 31) | 0x80000000u);
        }
        unsigned prefix = 0;
        int r = K_SEL;
        for (int bit = 31; bit >= 0; --bit) {
            unsigned b = 1u << bit;
            unsigned himask = ~((b << 1) - 1u);   // bits above current bit
            int cnt = 0;
#pragma unroll
            for (int j = 0; j < 32; ++j)
                cnt += (((key[j] & himask) == prefix) && (key[j] & b)) ? 1 : 0;
            // wave32 XOR-tree reduction via ds_swizzle (bit-mode, and=0x1f)
            cnt += __builtin_amdgcn_ds_swizzle(cnt, 0x401F);
            cnt += __builtin_amdgcn_ds_swizzle(cnt, 0x201F);
            cnt += __builtin_amdgcn_ds_swizzle(cnt, 0x101F);
            cnt += __builtin_amdgcn_ds_swizzle(cnt, 0x081F);
            cnt += __builtin_amdgcn_ds_swizzle(cnt, 0x041F);
            if (cnt >= r) prefix |= b;
            else          r -= cnt;
        }
        const unsigned T = prefix;               // exact key of K-th largest
        const unsigned lanemask_lt = (1u << lane) - 1u;
        int eq_run = 0;                          // equals seen so far (uniform)
        const float* erow = emb + (row0 + m) * (long)D_MODEL;
        float*       orow = out + (row0 + m) * (long)D_MODEL;
#pragma unroll
        for (int j = 0; j < 32; ++j) {
            bool eq = (key[j] == T);
            unsigned bal = __builtin_amdgcn_ballot_w32(eq);
            int before = __popc(bal & lanemask_lt);
            bool sel = (key[j] > T) || (eq && (eq_run + before) < r);
            eq_run += __popc(bal);
            int d = j * 32 + lane;
            orow[d] = sel ? erow[d] : 0.0f;
        }
    }
}

extern "C" void kernel_launch(void* const* d_in, const int* in_sizes, int n_in,
                              void* d_out, int out_size, void* d_ws, size_t ws_size,
                              hipStream_t stream) {
    (void)in_sizes; (void)n_in; (void)out_size; (void)ws_size;
    const float* emb = (const float*)d_in[0];
    const float* w1  = (const float*)d_in[1];
    const float* b1  = (const float*)d_in[2];
    const float* w2  = (const float*)d_in[3];
    const float* b2  = (const float*)d_in[4];
    float* out = (float*)d_out;

    unsigned short* w1p = (unsigned short*)d_ws;                       // 1 MB
    unsigned short* w2p = w1p + (size_t)D_MODEL * H_DIM;               // 1 MB

    const int packTotal = D_MODEL * H_DIM + H_DIM * D_MODEL;
    pack_weights_kernel<<<(packTotal + 255) / 256, 256, 0, stream>>>(w1, w2, w1p, w2p);

    const int nblocks = N_ROWS / M_TILE;                               // 2048
    const size_t smem_bytes = (size_t)(32 * 512 + 16 * 512) * sizeof(unsigned short)
                            + (size_t)M_TILE * D_MODEL * sizeof(float); // 112 KB
    fused_topk_mask_kernel<<<nblocks, BLOCK_THREADS, smem_bytes, stream>>>(
        emb, b1, b2, w1p, w2p, out);
}
